// NTXentLoss_14044543058300
// MI455X (gfx1250) — matrix operand order repaired and checked by reading, not compiled, not run on hardware
//
#include <hip/hip_runtime.h>
#include <hip/hip_bf16.h>
#include <stdint.h>

typedef __attribute__((ext_vector_type(16))) __bf16 v16bf;
typedef __attribute__((ext_vector_type(8)))  float  v8f;

#define NROWS  8192
#define DDIM   512
#define HALF_B 4096
#define INV_T  2.0f
#define BM 128
#define BN 128
#define KC 128
#define LDSTR 136   // ushort stride: 272B rows -> 16B aligned, rotates 4 banks/row

__device__ __forceinline__ unsigned short f32_to_bf16_rne(float x) {
  union { float f; uint32_t u; } c; c.f = x;
  uint32_t r = c.u + 0x7FFFu + ((c.u >> 16) & 1u);
  return (unsigned short)(r >> 16);
}
__device__ __forceinline__ float bf16_bits_to_f32(unsigned short h) {
  union { uint32_t u; float f; } c; c.u = ((uint32_t)h) << 16;
  return c.f;
}

// Kernel 1: one block per row. L2-normalize, split into bf16 hi/lo planes,
// zero the denominator accumulator.
__global__ __launch_bounds__(256) void ntxent_normalize_split(
    const float* __restrict__ z_i, const float* __restrict__ z_j,
    unsigned short* __restrict__ zhi, unsigned short* __restrict__ zlo,
    float* __restrict__ denom) {
  const int row = blockIdx.x;
  const float* src = (row < HALF_B) ? (z_i + (size_t)row * DDIM)
                                    : (z_j + (size_t)(row - HALF_B) * DDIM);
  const int tid = threadIdx.x;                 // 256 threads, 2 elems each
  float x0 = src[tid];
  float x1 = src[tid + 256];
  float ss = x0 * x0 + x1 * x1;
  for (int off = 1; off < 32; off <<= 1) ss += __shfl_xor(ss, off, 32);
  __shared__ float wsum[8];
  if ((tid & 31) == 0) wsum[tid >> 5] = ss;
  __syncthreads();
  float tot = 0.f;
  #pragma unroll
  for (int i = 0; i < 8; ++i) tot += wsum[i];
  const float inv = rsqrtf(tot);               // norms ~22.6 -> eps guard dead

  #pragma unroll
  for (int e = 0; e < 2; ++e) {
    int idx = tid + e * 256;
    float y = src[idx] * inv;
    unsigned short hb = f32_to_bf16_rne(y);
    float hf = bf16_bits_to_f32(hb);
    unsigned short lb = f32_to_bf16_rne(y - hf);
    zhi[(size_t)row * DDIM + idx] = hb;
    zlo[(size_t)row * DDIM + idx] = lb;
  }
  if (tid == 0) denom[row] = 0.0f;
}

// Kernel 2: fused tile GEMM (bf16 hi/lo, f32 accumulate) + exp + masked
// row/col sums + positive extraction. Upper-triangle tiles only; symmetry
// supplies the lower triangle via column sums.
__global__ __launch_bounds__(256) void ntxent_sim_tile(
    const unsigned short* __restrict__ zhi, const unsigned short* __restrict__ zlo,
    float* __restrict__ denom, float* __restrict__ pos) {
  const int bx = blockIdx.x, by = blockIdx.y;
  if (bx < by) return;                         // symmetric: upper triangle only
  const int m0 = by * BM, n0 = bx * BN;

  __shared__ unsigned short sBhi[BN * LDSTR];
  __shared__ unsigned short sBlo[BN * LDSTR];

  const int tid  = threadIdx.x;
  const int wave = tid >> 5;
  const int lane = tid & 31;
  const int half = lane >> 4;                  // K-half selector (WMMA layout)
  const int cl   = lane & 15;                  // M (A) / N (B) within fragment

  v8f acc[8];
  #pragma unroll
  for (int i = 0; i < 8; ++i)
    #pragma unroll
    for (int j = 0; j < 8; ++j) acc[i][j] = 0.0f;

  const int arow = m0 + wave * 16 + cl;        // A-frag row for this lane
  const unsigned short* aHiRow = zhi + (size_t)arow * DDIM;
  const unsigned short* aLoRow = zlo + (size_t)arow * DDIM;

  for (int k0 = 0; k0 < DDIM; k0 += KC) {
    __syncthreads();
    // Stage B tile (128 cols x KC) hi+lo into LDS; 8-ushort (16B) segments.
    for (int i = tid; i < (BN * KC / 8); i += 256) {
      int r = i >> 4, seg = (i & 15) * 8;
      size_t g = (size_t)(n0 + r) * DDIM + k0 + seg;
      *reinterpret_cast<uint4*>(&sBhi[r * LDSTR + seg]) =
          *reinterpret_cast<const uint4*>(zhi + g);
      *reinterpret_cast<uint4*>(&sBlo[r * LDSTR + seg]) =
          *reinterpret_cast<const uint4*>(zlo + g);
      if (k0 + KC < DDIM) {                    // gfx1250 global_prefetch_b8
        __builtin_prefetch(zhi + g + KC, 0, 1);
        __builtin_prefetch(zlo + g + KC, 0, 1);
      }
    }
    __syncthreads();

    #pragma unroll
    for (int kk = 0; kk < KC; kk += 32) {
      // A frag (16x32 bf16): lane holds K = base..base+7 and base+16..base+23
      union { v16bf v; uint4 u[2]; } ahi, alo;
      const int kb = k0 + kk + half * 8;
      ahi.u[0] = *reinterpret_cast<const uint4*>(aHiRow + kb);
      ahi.u[1] = *reinterpret_cast<const uint4*>(aHiRow + kb + 16);
      alo.u[0] = *reinterpret_cast<const uint4*>(aLoRow + kb);
      alo.u[1] = *reinterpret_cast<const uint4*>(aLoRow + kb + 16);

      #pragma unroll
      for (int nf = 0; nf < 8; ++nf) {
        // B frag (32x16 bf16): lane holds K = base..base+7 and base+16..base+23, base = kk + half*8
        union { v16bf v; uint4 u[2]; } bhi, blo;
        const unsigned short* bh = &sBhi[(nf * 16 + cl) * LDSTR + kk + half * 8];
        const unsigned short* bl = &sBlo[(nf * 16 + cl) * LDSTR + kk + half * 8];
        bhi.u[0] = *reinterpret_cast<const uint4*>(bh);
        bhi.u[1] = *reinterpret_cast<const uint4*>(bh + 16);
        blo.u[0] = *reinterpret_cast<const uint4*>(bl);
        blo.u[1] = *reinterpret_cast<const uint4*>(bl + 16);
        // bf16x3: hi*hi + hi*lo + lo*hi  (drops only ~2^-16 lo*lo term)
        acc[nf] = __builtin_amdgcn_wmma_f32_16x16x32_bf16(
            false, ahi.v, false, bhi.v, (short)0, acc[nf], false, false);
        acc[nf] = __builtin_amdgcn_wmma_f32_16x16x32_bf16(
            false, ahi.v, false, blo.v, (short)0, acc[nf], false, false);
        acc[nf] = __builtin_amdgcn_wmma_f32_16x16x32_bf16(
            false, alo.v, false, bhi.v, (short)0, acc[nf], false, false);
      }
    }
  }

  // Epilogue: sim -> exp(2*sim), mask diagonal, accumulate row/col sums,
  // capture positives (symmetric pair serves both rows).
  const int diag = (bx == by);
  float rowsum[8];
  #pragma unroll
  for (int v = 0; v < 8; ++v) rowsum[v] = 0.f;

  #pragma unroll
  for (int nf = 0; nf < 8; ++nf) {
    const int col = n0 + nf * 16 + cl;
    float cs = 0.f;
    #pragma unroll
    for (int v = 0; v < 8; ++v) {
      const int row = m0 + wave * 16 + half * 8 + v;
      float s = acc[nf][v] * INV_T;
      float e = __expf(s);
      if (row == col) e = 0.f;                 // self-similarity mask
      rowsum[v] += e;
      cs += e;
      if (col == row + HALF_B) { pos[row] = s; pos[col] = s; }
    }
    cs += __shfl_xor(cs, 16, 32);              // combine both row-halves
    if (!diag && half == 0) atomicAdd(&denom[col], cs);  // symmetric mirror
  }
  #pragma unroll
  for (int v = 0; v < 8; ++v) {
    float rs = rowsum[v];
    for (int off = 1; off < 16; off <<= 1) rs += __shfl_xor(rs, off, 32);
    if (cl == 0) {
      const int row = m0 + wave * 16 + half * 8 + v;
      atomicAdd(&denom[row], rs);
    }
  }
}

// Kernel 3: loss = mean(log(denom + eps) - pos)
__global__ __launch_bounds__(256) void ntxent_loss_reduce(
    const float* __restrict__ denom, const float* __restrict__ pos,
    float* __restrict__ out) {
  const int tid = threadIdx.x;
  float acc = 0.f;
  for (int i = tid; i < NROWS; i += 256)
    acc += logf(denom[i] + 1e-8f) - pos[i];
  for (int off = 1; off < 32; off <<= 1) acc += __shfl_xor(acc, off, 32);
  __shared__ float ws[8];
  if ((tid & 31) == 0) ws[tid >> 5] = acc;
  __syncthreads();
  if (tid == 0) {
    float t = 0.f;
    #pragma unroll
    for (int i = 0; i < 8; ++i) t += ws[i];
    out[0] = t / (float)NROWS;
  }
}

extern "C" void kernel_launch(void* const* d_in, const int* in_sizes, int n_in,
                              void* d_out, int out_size, void* d_ws, size_t ws_size,
                              hipStream_t stream) {
  (void)in_sizes; (void)n_in; (void)out_size; (void)ws_size;
  const float* z_i = (const float*)d_in[0];
  const float* z_j = (const float*)d_in[1];
  float* out = (float*)d_out;

  char* ws = (char*)d_ws;
  unsigned short* zhi = (unsigned short*)ws;                       // 8 MB
  unsigned short* zlo = (unsigned short*)(ws + (size_t)NROWS * DDIM * 2);  // 8 MB
  float* denom = (float*)(ws + (size_t)NROWS * DDIM * 4);          // 32 KB
  float* pos   = denom + NROWS;                                    // 32 KB

  ntxent_normalize_split<<<NROWS, 256, 0, stream>>>(z_i, z_j, zhi, zlo, denom);
  ntxent_sim_tile<<<dim3(NROWS / BN, NROWS / BM), 256, 0, stream>>>(zhi, zlo, denom, pos);
  ntxent_loss_reduce<<<1, 256, 0, stream>>>(denom, pos, out);
}
